// MLP_Param_33741263078240
// MI455X (gfx1250) — matrix operand order repaired and checked by reading, not compiled, not run on hardware
//
#include <hip/hip_runtime.h>

typedef __attribute__((ext_vector_type(16))) __bf16 v16bf;
typedef __attribute__((ext_vector_type(8)))  __bf16 v8bf;
typedef __attribute__((ext_vector_type(8)))  float  v8f;
typedef __attribute__((ext_vector_type(4)))  float  f32x4;

#define Nn 4096
#define Dd 64
#define Ii 512
#define Oo 512
#define KTOT (Dd * Ii)        // 32768
#define KB 128                // K per LDS stage
#define NKB (KTOT / KB)       // 256 regular stages
#define STAGES (NKB + 1)      // +1 bias stage (K pad 64 -> 128 with zeros)
#define BM 128
#define BN 64
#define LDA 136               // padded bf16 elements per LDS row (272 B)

// Build one K-stage in LDS.
//  As[m][k]  = var[n0+m][d(kb)] * x[n0+m][i0(kb)+k]   (bf16), or [var | 0] on bias stage
//  Bs[o][k]  = W_flat[kb*128 + k][o0+o]               (bf16, transposed), or [b_1 ; 0]
__device__ __forceinline__ void stage_load(
    int kb, __bf16 (*As)[LDA], __bf16 (*Bs)[LDA],
    const float* __restrict__ x, const float* __restrict__ var,
    const float* __restrict__ W, const float* __restrict__ bias,
    int n0, int o0, int tid)
{
  // ---------------- A tile: 128 rows x 128 k, 2 threads/row, 64 floats/thread
  {
    const int row  = tid >> 1;           // 0..127
    const int colc = (tid & 1) * 64;     // 0 or 64
    const int m    = n0 + row;
    if (kb < NKB) {
      const int d  = kb >> 2;            // 128-wide K block stays inside one d
      const int i0 = (kb & 3) * KB;
      const float s = var[(size_t)m * Dd + d];
      const f32x4* xp = (const f32x4*)(x + (size_t)m * Ii + i0 + colc);
#pragma unroll
      for (int v = 0; v < 16; ++v) {
        f32x4 f = xp[v];
#pragma unroll
        for (int e = 0; e < 4; ++e)
          As[row][colc + v * 4 + e] = (__bf16)(f[e] * s);
      }
    } else {                             // bias stage: A = [var(64) | zeros(64)]
      if (colc == 0) {
        const f32x4* vp = (const f32x4*)(var + (size_t)m * Dd);
#pragma unroll
        for (int v = 0; v < 16; ++v) {
          f32x4 f = vp[v];
#pragma unroll
          for (int e = 0; e < 4; ++e)
            As[row][v * 4 + e] = (__bf16)f[e];
        }
      } else {
#pragma unroll
        for (int v = 0; v < 64; ++v) As[row][64 + v] = (__bf16)0.0f;
      }
    }
  }
  // ---------------- B tile (transposed into LDS): 128 k x 64 o
  // thread owns a 4k x 8o sub-block: coalesced global float4 reads,
  // contiguous 4-element bf16 scatters per output column.
  {
    const int og    = tid & 7;           // o-group: 8 groups of 8
    const int kg    = tid >> 3;          // 0..31
    const int kbase = kg * 4;
    const int ob    = og * 8;
    float vals[4][8];
    if (kb < NKB) {
      const size_t k0 = (size_t)kb * KB;
#pragma unroll
      for (int r = 0; r < 4; ++r) {
        const f32x4* wp = (const f32x4*)(W + (k0 + kbase + r) * (size_t)Oo + o0 + ob);
        f32x4 a = wp[0], b4 = wp[1];
#pragma unroll
        for (int e = 0; e < 4; ++e) { vals[r][e] = a[e]; vals[r][4 + e] = b4[e]; }
      }
      // Prefetch the W rows we will need two stages from now (covers L2 latency).
      if (kb + 2 < NKB)
        __builtin_prefetch(W + (k0 + 2 * KB + kbase) * (size_t)Oo + o0 + ob, 0, 0);
    } else {                             // bias stage: B = [b_1(64 rows) ; zeros]
#pragma unroll
      for (int r = 0; r < 4; ++r) {
        if (kbase + r < Dd) {
          const f32x4* bp = (const f32x4*)(bias + (size_t)(kbase + r) * Oo + o0 + ob);
          f32x4 a = bp[0], b4 = bp[1];
#pragma unroll
          for (int e = 0; e < 4; ++e) { vals[r][e] = a[e]; vals[r][4 + e] = b4[e]; }
        } else {
#pragma unroll
          for (int e = 0; e < 8; ++e) vals[r][e] = 0.0f;
        }
      }
    }
#pragma unroll
    for (int oo = 0; oo < 8; ++oo) {
      __bf16* p = &Bs[ob + oo][kbase];
      p[0] = (__bf16)vals[0][oo];
      p[1] = (__bf16)vals[1][oo];
      p[2] = (__bf16)vals[2][oo];
      p[3] = (__bf16)vals[3][oo];
    }
  }
}

__global__ __launch_bounds__(256) void kan_mlp_wmma_kernel(
    const float* __restrict__ x, const float* __restrict__ var,
    const float* __restrict__ W, const float* __restrict__ bias,
    float* __restrict__ out)
{
  __shared__ __bf16 Asm[2][BM][LDA];   // 68 KB
  __shared__ __bf16 Bsm[2][BN][LDA];   // 34 KB (transposed: [o][k])

  const int tid  = threadIdx.x;
  const int n0   = blockIdx.y * BM;
  const int o0   = blockIdx.x * BN;

  const int lane = tid & 31;
  const int wid  = tid >> 5;    // 0..7
  const int wm   = wid >> 1;    // 0..3  (32-row M slice)
  const int wn   = wid & 1;     // 0..1  (32-col N slice)
  const int c    = lane & 15;   // A row / B column within fragment
  const int h    = lane >> 4;   // half-wave select

  v8f acc00 = {}, acc01 = {}, acc10 = {}, acc11 = {};

  stage_load(0, Asm[0], Bsm[0], x, var, W, bias, n0, o0, tid);
  __syncthreads();

  for (int kb = 0; kb < STAGES; ++kb) {
    const int buf = kb & 1;
    if (kb + 1 < STAGES)
      stage_load(kb + 1, Asm[buf ^ 1], Bsm[buf ^ 1], x, var, W, bias, n0, o0, tid);

#pragma unroll
    for (int kk = 0; kk < 4; ++kk) {
      const int ks = kk * 32;

      // A fragments (16x32 bf16): lanes 0-15 -> K 0-7,16-23 ; lanes 16-31 -> K 8-15,24-31
      v16bf a0, a1;
      {
        const v8bf lo = *(const v8bf*)&Asm[buf][wm * 32 + c][ks + h * 8];
        const v8bf hi = *(const v8bf*)&Asm[buf][wm * 32 + c][ks + 16 + h * 8];
        a0 = __builtin_shufflevector(lo, hi,
            0, 1, 2, 3, 4, 5, 6, 7, 8, 9, 10, 11, 12, 13, 14, 15);
      }
      {
        const v8bf lo = *(const v8bf*)&Asm[buf][wm * 32 + 16 + c][ks + h * 8];
        const v8bf hi = *(const v8bf*)&Asm[buf][wm * 32 + 16 + c][ks + 16 + h * 8];
        a1 = __builtin_shufflevector(lo, hi,
            0, 1, 2, 3, 4, 5, 6, 7, 8, 9, 10, 11, 12, 13, 14, 15);
      }

      // B fragments (32x16 bf16): lanes 0-15 -> K 0-7,16-23 ; lanes 16-31 -> K 8-15,24-31
      v16bf b0, b1;
      {
        const v8bf lo = *(const v8bf*)&Bsm[buf][wn * 32 + c][ks + h * 8];
        const v8bf hi = *(const v8bf*)&Bsm[buf][wn * 32 + c][ks + 16 + h * 8];
        b0 = __builtin_shufflevector(lo, hi,
            0, 1, 2, 3, 4, 5, 6, 7, 8, 9, 10, 11, 12, 13, 14, 15);
      }
      {
        const v8bf lo = *(const v8bf*)&Bsm[buf][wn * 32 + 16 + c][ks + h * 8];
        const v8bf hi = *(const v8bf*)&Bsm[buf][wn * 32 + 16 + c][ks + 16 + h * 8];
        b1 = __builtin_shufflevector(lo, hi,
            0, 1, 2, 3, 4, 5, 6, 7, 8, 9, 10, 11, 12, 13, 14, 15);
      }

      // 4 WMMAs, each fragment reused twice.
      acc00 = __builtin_amdgcn_wmma_f32_16x16x32_bf16(
          false, a0, false, b0, (short)0, acc00, false, false);
      acc01 = __builtin_amdgcn_wmma_f32_16x16x32_bf16(
          false, a0, false, b1, (short)0, acc01, false, false);
      acc10 = __builtin_amdgcn_wmma_f32_16x16x32_bf16(
          false, a1, false, b0, (short)0, acc10, false, false);
      acc11 = __builtin_amdgcn_wmma_f32_16x16x32_bf16(
          false, a1, false, b1, (short)0, acc11, false, false);
    }
    __syncthreads();
  }

  // Epilogue: C/D layout — VGPR j: lanes 0-15 M=j, lanes 16-31 M=8+j; N=lane&15
  const int orow = n0 + wm * 32 + h * 8;
  const int ocol = o0 + wn * 32 + c;
#pragma unroll
  for (int j = 0; j < 8; ++j) {
    out[(size_t)(orow + j) * Oo + ocol]           = acc00[j];
    out[(size_t)(orow + j) * Oo + ocol + 16]      = acc01[j];
    out[(size_t)(orow + 16 + j) * Oo + ocol]      = acc10[j];
    out[(size_t)(orow + 16 + j) * Oo + ocol + 16] = acc11[j];
  }
}

extern "C" void kernel_launch(void* const* d_in, const int* in_sizes, int n_in,
                              void* d_out, int out_size, void* d_ws, size_t ws_size,
                              hipStream_t stream) {
  (void)in_sizes; (void)n_in; (void)out_size; (void)d_ws; (void)ws_size;
  const float* x    = (const float*)d_in[0];   // [4096, 512]
  const float* var  = (const float*)d_in[1];   // [4096, 64]
  const float* W    = (const float*)d_in[2];   // [64, 512, 512] == W_flat[32768, 512]
  const float* bias = (const float*)d_in[3];   // [64, 512]
  float* out = (float*)d_out;                  // [4096, 512]

  dim3 grid(Oo / BN, Nn / BM);                 // (8, 32)
  kan_mlp_wmma_kernel<<<grid, dim3(256), 0, stream>>>(x, var, W, bias, out);
}
